// Step3TextMoEMLP_4990751998317
// MI455X (gfx1250) — matrix operand order repaired and checked by reading, not compiled, not run on hardware
//
#include <hip/hip_runtime.h>
#include <math.h>

// Problem constants (match reference)
#define TOK 2048   // tokens T
#define HID 2048   // hidden H
#define NE  8      // experts E
#define IM  1024   // intermediate I

typedef __attribute__((ext_vector_type(4)))  __bf16 v4bf;
typedef __attribute__((ext_vector_type(8)))  __bf16 v8bf;
typedef __attribute__((ext_vector_type(16))) __bf16 v16bf;
typedef __attribute__((ext_vector_type(8)))  float  v8f;

#define LDSK 40  // row stride (bf16 elems) for [.][k] tiles: 80 B, 16B-aligned

static __device__ __forceinline__ v16bf cat16(v8bf lo, v8bf hi) {
  return __builtin_shufflevector(lo, hi, 0,1,2,3,4,5,6,7,8,9,10,11,12,13,14,15);
}

// ---------------------------------------------------------------------------
// Kernel 1: router. One wave per token: logits = h . gate_w, softmax, top-2,
// renormalize, atomic scatter into per-expert (token, weight) lists.
// ---------------------------------------------------------------------------
__global__ __launch_bounds__(256) void router_kernel(
    const float* __restrict__ hidden, const float* __restrict__ gate_w,
    int* __restrict__ counts, int* __restrict__ tidx, float* __restrict__ tw) {
  const int wave = threadIdx.x >> 5;
  const int lane = threadIdx.x & 31;
  const int t = blockIdx.x * 8 + wave;
  if (t >= TOK) return;

  float acc[NE];
#pragma unroll
  for (int e = 0; e < NE; ++e) acc[e] = 0.f;

  const float* hrow = hidden + (size_t)t * HID;
  for (int h = lane; h < HID; h += 32) {
    const float hv = hrow[h];
    const float* gw = gate_w + (size_t)h * NE;
#pragma unroll
    for (int e = 0; e < NE; ++e) acc[e] += hv * gw[e];
  }
#pragma unroll
  for (int e = 0; e < NE; ++e) {
#pragma unroll
    for (int off = 16; off > 0; off >>= 1)
      acc[e] += __shfl_xor(acc[e], off, 32);
  }

  if (lane == 0) {
    float mx = acc[0];
#pragma unroll
    for (int e = 1; e < NE; ++e) mx = fmaxf(mx, acc[e]);
    float p[NE];
    float s = 0.f;
#pragma unroll
    for (int e = 0; e < NE; ++e) { p[e] = __expf(acc[e] - mx); s += p[e]; }
    const float inv_s = 1.f / s;
#pragma unroll
    for (int e = 0; e < NE; ++e) p[e] *= inv_s;
    int i0 = 0;
#pragma unroll
    for (int e = 1; e < NE; ++e) if (p[e] > p[i0]) i0 = e;
    int i1 = (i0 == 0) ? 1 : 0;
#pragma unroll
    for (int e = 0; e < NE; ++e) if (e != i0 && p[e] > p[i1]) i1 = e;
    float w0 = p[i0], w1 = p[i1];
    const float inv = 1.f / (w0 + w1);
    w0 *= inv; w1 *= inv;

    int pos0 = atomicAdd(&counts[i0], 1);
    tidx[i0 * TOK + pos0] = t;
    tw[i0 * TOK + pos0] = w0;
    int pos1 = atomicAdd(&counts[i1], 1);
    tidx[i1 * TOK + pos1] = t;
    tw[i1 * TOK + pos1] = w1;
  }
}

// ---------------------------------------------------------------------------
// Kernel 2: gathered gate+up GEMM + SiLU. Block = 256 threads = 8 waves (2x4),
// tile M=64 x N=64, K-loop over H in 32s. Each wave owns two stacked 16x16
// M-subtiles that share its B fragments -> 4 bf16 WMMAs per wave per K-step.
// ---------------------------------------------------------------------------
__global__ __launch_bounds__(256) void gateup_silu_kernel(
    const float* __restrict__ hidden, const float* __restrict__ w13,
    const int* __restrict__ counts, const int* __restrict__ tidx,
    __bf16* __restrict__ act) {
  const int e = blockIdx.z;
  const int cnt = counts[e];
  const int m0 = blockIdx.y * 64;
  if (m0 >= cnt) return;  // uniform exit: EXEC all-ones for WMMA
  const int i0 = blockIdx.x * 64;

  __shared__ __attribute__((aligned(16))) __bf16 As[64][LDSK];   // [row][k]
  __shared__ __attribute__((aligned(16))) __bf16 BgT[64][LDSK];  // [n][k]
  __shared__ __attribute__((aligned(16))) __bf16 BuT[64][LDSK];  // [n][k]

  const int tid = threadIdx.x;
  const int wave = tid >> 5, lane = tid & 31;
  const int wm = wave >> 2, wn = wave & 3;   // 2x4 wave grid
  const int hi = lane >> 4, lm = lane & 15;

  // --- A staging: thread -> 2 (row, 4-elem chunk) units. Invalid rows read
  // token 0 (their WMMA outputs are discarded by the store guard).
  const int ar0 = tid >> 3, ar1 = ar0 + 32;
  const int ac = (tid & 7) * 4;
  const int s0 = m0 + ar0, s1 = m0 + ar1;
  const int t0 = (s0 < cnt) ? tidx[e * TOK + s0] : 0;
  const int t1 = (s1 < cnt) ? tidx[e * TOK + s1] : 0;
  const float* arow0 = hidden + (size_t)t0 * HID + ac;
  const float* arow1 = hidden + (size_t)t1 * HID + ac;

  // --- B staging: thread -> one 4k x 4n micro-tile of gate or up.
  const int bmat = tid >> 7;            // 0 = gate, 1 = up
  const int bk4 = (tid & 127) >> 4;     // 0..7  -> k = bk4*4
  const int bn4 = tid & 15;             // 0..15 -> n = bn4*4
  const float* bbase =
      w13 + ((size_t)e * HID) * (2 * IM) + (size_t)bmat * IM + i0 + bn4 * 4;
  __bf16* bdst = (bmat ? &BuT[0][0] : &BgT[0][0]) + (bn4 * 4) * LDSK + bk4 * 4;

  v8f accg[2] = {v8f{}, v8f{}};
  v8f accu[2] = {v8f{}, v8f{}};

  for (int k0 = 0; k0 < HID; k0 += 32) {
    // A: two b128 loads -> packed bf16 -> two b64 LDS stores
    {
      const float4 av0 = *(const float4*)(arow0 + k0);
      const float4 av1 = *(const float4*)(arow1 + k0);
      v4bf a4;
      a4.x = (__bf16)av0.x; a4.y = (__bf16)av0.y;
      a4.z = (__bf16)av0.z; a4.w = (__bf16)av0.w;
      *(v4bf*)&As[ar0][ac] = a4;
      a4.x = (__bf16)av1.x; a4.y = (__bf16)av1.y;
      a4.z = (__bf16)av1.z; a4.w = (__bf16)av1.w;
      *(v4bf*)&As[ar1][ac] = a4;
    }

    // B: 4 b128 row loads, register transpose, 4 b64 column stores
    const float* bp = bbase + (size_t)(k0 + bk4 * 4) * (2 * IM);
    const float4 r0 = *(const float4*)(bp);
    const float4 r1 = *(const float4*)(bp + 2 * IM);
    const float4 r2 = *(const float4*)(bp + 4 * IM);
    const float4 r3 = *(const float4*)(bp + 6 * IM);
    v4bf c0, c1, c2, c3;
    c0.x = (__bf16)r0.x; c0.y = (__bf16)r1.x; c0.z = (__bf16)r2.x; c0.w = (__bf16)r3.x;
    c1.x = (__bf16)r0.y; c1.y = (__bf16)r1.y; c1.z = (__bf16)r2.y; c1.w = (__bf16)r3.y;
    c2.x = (__bf16)r0.z; c2.y = (__bf16)r1.z; c2.z = (__bf16)r2.z; c2.w = (__bf16)r3.z;
    c3.x = (__bf16)r0.w; c3.y = (__bf16)r1.w; c3.z = (__bf16)r2.w; c3.w = (__bf16)r3.w;
    *(v4bf*)(bdst)            = c0;
    *(v4bf*)(bdst + LDSK)     = c1;
    *(v4bf*)(bdst + 2 * LDSK) = c2;
    *(v4bf*)(bdst + 3 * LDSK) = c3;

    if (k0 + 32 < HID)
      __builtin_prefetch(bbase + (size_t)(k0 + 32 + bk4 * 4) * (2 * IM), 0, 1);
    __syncthreads();

    // B fragments (shared by both M-subtiles of this wave)
    const int bn = wn * 16 + lm;
    const v16bf bg = cat16(*(const v8bf*)&BgT[bn][hi * 8],
                           *(const v8bf*)&BgT[bn][16 + hi * 8]);
    const v16bf bu = cat16(*(const v8bf*)&BuT[bn][hi * 8],
                           *(const v8bf*)&BuT[bn][16 + hi * 8]);
    // Two A fragments: rows wm*32 + {0,16} + lm
#pragma unroll
    for (int st = 0; st < 2; ++st) {
      const int am = wm * 32 + st * 16 + lm;
      const v16bf a = cat16(*(const v8bf*)&As[am][hi * 8],
                            *(const v8bf*)&As[am][16 + hi * 8]);
      accg[st] = __builtin_amdgcn_wmma_f32_16x16x32_bf16(
          false, a, false, bg, (short)0, accg[st], false, false);
      accu[st] = __builtin_amdgcn_wmma_f32_16x16x32_bf16(
          false, a, false, bu, (short)0, accu[st], false, false);
    }
    __syncthreads();
  }

  // Epilogue: act = silu(g) * u, bf16 to workspace
#pragma unroll
  for (int st = 0; st < 2; ++st) {
#pragma unroll
    for (int r = 0; r < 8; ++r) {
      const int lrow = wm * 32 + st * 16 + hi * 8 + r;
      const int slot = m0 + lrow;
      if (slot < cnt) {
        const float g = accg[st][r], u = accu[st][r];
        const float sv = g / (1.f + __expf(-g));
        act[((size_t)e * TOK + slot) * IM + i0 + wn * 16 + lm] = (__bf16)(sv * u);
      }
    }
  }
}

// ---------------------------------------------------------------------------
// Kernel 3: down projection + weighted combine. Tile M=64 x N=64. A-tile
// (bf16 act) staged with CDNA5 async global->LDS copies (ASYNCcnt); B-tile
// converts w2 fp32 -> bf16 vectorized. Epilogue scales by the router weight
// and atomically accumulates into the pre-zeroed output (K=2 combine).
// ---------------------------------------------------------------------------
__global__ __launch_bounds__(256) void down_combine_kernel(
    const __bf16* __restrict__ act, const float* __restrict__ w2,
    const int* __restrict__ counts, const int* __restrict__ tidx,
    const float* __restrict__ tw, float* __restrict__ out) {
  const int e = blockIdx.z;
  const int cnt = counts[e];
  const int m0 = blockIdx.y * 64;
  if (m0 >= cnt) return;
  const int h0 = blockIdx.x * 64;

  __shared__ __attribute__((aligned(16))) __bf16 As[64][LDSK];  // [row][k]
  __shared__ __attribute__((aligned(16))) __bf16 BT[64][LDSK];  // [n][k]
  __shared__ int stok[64];
  __shared__ float sw[64];

  const int tid = threadIdx.x;
  const int wave = tid >> 5, lane = tid & 31;
  const int wm = wave >> 2, wn = wave & 3;
  const int hi = lane >> 4, lm = lane & 15;

  if (tid < 64) {
    const int slot = m0 + tid;
    stok[tid] = (slot < cnt) ? tidx[e * TOK + slot] : 0;
    sw[tid] = (slot < cnt) ? tw[e * TOK + slot] : 0.f;
  }
  __syncthreads();

  // --- A staging via async copy: thread -> 2 (row, 8-byte chunk) units
  const int ar0 = tid >> 3, ar1 = ar0 + 32;
  const int ac = (tid & 7) * 4;  // 4 bf16 = 8 bytes
  int s0 = m0 + ar0, s1 = m0 + ar1;
  if (s0 >= cnt) s0 = cnt - 1;  // clamp; extra rows' outputs are discarded
  if (s1 >= cnt) s1 = cnt - 1;
  const __bf16* arow0 = act + ((size_t)e * TOK + s0) * IM + ac;
  const __bf16* arow1 = act + ((size_t)e * TOK + s1) * IM + ac;
  const unsigned alds0 = (unsigned)(uintptr_t)&As[ar0][ac];
  const unsigned alds1 = (unsigned)(uintptr_t)&As[ar1][ac];

  // --- B staging: thread -> 4k x 2n micro-tile
  const int bk4 = tid >> 5;            // 0..7 -> k = bk4*4
  const int bn2 = (tid & 31) * 2;      // 0..62
  const float* bbase = w2 + ((size_t)e * IM) * HID + h0 + bn2;
  __bf16* bdst = &BT[bn2][bk4 * 4];

  v8f acc[2] = {v8f{}, v8f{}};

  for (int k0 = 0; k0 < IM; k0 += 32) {
    // A: async global->LDS b64 (per-lane global addr, per-lane LDS addr)
    {
      const unsigned long long ga0 = (unsigned long long)(uintptr_t)(arow0 + k0);
      const unsigned long long ga1 = (unsigned long long)(uintptr_t)(arow1 + k0);
      asm volatile("global_load_async_to_lds_b64 %0, %1, off"
                   :: "v"(alds0), "v"(ga0) : "memory");
      asm volatile("global_load_async_to_lds_b64 %0, %1, off"
                   :: "v"(alds1), "v"(ga1) : "memory");
    }

    // B: 4 b64 row loads, register transpose, 2 b64 column stores
    const float* bp = bbase + (size_t)(k0 + bk4 * 4) * HID;
    const float2 r0 = *(const float2*)(bp);
    const float2 r1 = *(const float2*)(bp + HID);
    const float2 r2 = *(const float2*)(bp + 2 * HID);
    const float2 r3 = *(const float2*)(bp + 3 * HID);
    v4bf c0, c1;
    c0.x = (__bf16)r0.x; c0.y = (__bf16)r1.x; c0.z = (__bf16)r2.x; c0.w = (__bf16)r3.x;
    c1.x = (__bf16)r0.y; c1.y = (__bf16)r1.y; c1.z = (__bf16)r2.y; c1.w = (__bf16)r3.y;
    *(v4bf*)(bdst)        = c0;
    *(v4bf*)(bdst + LDSK) = c1;

    if (k0 + 32 < IM)
      __builtin_prefetch(bbase + (size_t)(k0 + 32 + bk4 * 4) * HID, 0, 1);

    asm volatile("s_wait_asynccnt 0x0" ::: "memory");
    __syncthreads();

    const int bn = wn * 16 + lm;
    const v16bf b = cat16(*(const v8bf*)&BT[bn][hi * 8],
                          *(const v8bf*)&BT[bn][16 + hi * 8]);
#pragma unroll
    for (int st = 0; st < 2; ++st) {
      const int am = wm * 32 + st * 16 + lm;
      const v16bf a = cat16(*(const v8bf*)&As[am][hi * 8],
                            *(const v8bf*)&As[am][16 + hi * 8]);
      acc[st] = __builtin_amdgcn_wmma_f32_16x16x32_bf16(
          false, a, false, b, (short)0, acc[st], false, false);
    }
    __syncthreads();
  }

  const int n = h0 + wn * 16 + lm;
#pragma unroll
  for (int st = 0; st < 2; ++st) {
#pragma unroll
    for (int r = 0; r < 8; ++r) {
      const int lrow = wm * 32 + st * 16 + hi * 8 + r;
      const int slot = m0 + lrow;
      if (slot < cnt) {
        const int tok = stok[lrow];
        const float w = sw[lrow];
        unsafeAtomicAdd(&out[(size_t)tok * HID + n], w * acc[st][r]);
      }
    }
  }
}

// ---------------------------------------------------------------------------
// Launch: memset counts + output, then router -> gate/up+silu -> down+combine.
// All async on `stream` (graph-capture safe).
// ---------------------------------------------------------------------------
extern "C" void kernel_launch(void* const* d_in, const int* in_sizes, int n_in,
                              void* d_out, int out_size, void* d_ws, size_t ws_size,
                              hipStream_t stream) {
  const float* hidden = (const float*)d_in[0];
  const float* gate_w = (const float*)d_in[1];
  const float* w13    = (const float*)d_in[2];
  const float* w2     = (const float*)d_in[3];
  float* out = (float*)d_out;

  // Workspace layout
  char* ws = (char*)d_ws;
  int*    counts = (int*)ws;                                   // 256 B (padded)
  int*    tidx   = (int*)(ws + 256);                           // E*T ints
  float*  tw     = (float*)(ws + 256 + NE * TOK * 4);          // E*T floats
  __bf16* act    = (__bf16*)(ws + 256 + NE * TOK * 8);         // E*T*I bf16

  hipMemsetAsync(counts, 0, 256, stream);
  hipMemsetAsync(d_out, 0, (size_t)TOK * HID * sizeof(float), stream);

  router_kernel<<<TOK / 8, 256, 0, stream>>>(hidden, gate_w, counts, tidx, tw);

  gateup_silu_kernel<<<dim3(IM / 64, TOK / 64, NE), 256, 0, stream>>>(
      hidden, w13, counts, tidx, act);

  down_combine_kernel<<<dim3(HID / 64, TOK / 64, NE), 256, 0, stream>>>(
      act, w2, counts, tidx, tw, out);
}
